// LTCSequenceODE_13125420057030
// MI455X (gfx1250) — matrix operand order repaired and checked by reading, not compiled, not run on hardware
//
#include <hip/hip_runtime.h>
#include <hip/hip_bf16.h>
#include <math.h>

// MI455X / gfx1250, wave32. WMMA 16x16x32 bf16 -> f32.
typedef __attribute__((ext_vector_type(16))) __bf16 v16bf;
typedef __attribute__((ext_vector_type(8)))  float  v8f;
typedef __attribute__((ext_vector_type(4)))  unsigned int v4u;
typedef __attribute__((ext_vector_type(8)))  int v8i;
typedef __attribute__((ext_vector_type(4)))  int v4i;

#define HIDDEN 512
#define INPUT  256
#define T_IN   4096
#define T_SAVE 1024
#define NSTEP  (T_SAVE - 1)        // 1023
#define NSTAGE (4 * NSTEP)         // 4092
#define NSTAGE_P 4096              // padded to 16
#define NWG    4                   // cooperating workgroups in ODE kernel

// ---- workspace layout (bytes) ----
// 0        : unsigned counter (sync)
// 1024     : xbuf  (2 * 512 f32 ping-pong)
// 8192     : wpack (2 * 512 * 512 bf16, WMMA-fragment order) = 1 MB
// 8192+1M  : Ztau  (4096 * 512 f32) = 8 MB
// +8M      : Zf    (4096 * 512 f32) = 8 MB
#define WS_XBUF   1024
#define WS_WPACK  8192
#define WS_ZTAU   (WS_WPACK + 2*512*512*2)
#define WS_ZF     (WS_ZTAU + NSTAGE_P*HIDDEN*4)

// A-fragment K index for 16-bit A 16x32 (ISA 7.12.2): lane<16: e0..7->K0..7,
// e8..15->K16..23 ; lane>=16: e0..7->K8..15, e8..15->K24..31.
__device__ __forceinline__ int a_kidx(int e, int hi) {
    return e + 8 * hi + ((e >= 8) ? 8 : 0);
}

#if __has_builtin(__builtin_amdgcn_tensor_load_to_lds)
// Tensor Data Mover: DMA a contiguous 131072-byte block (16384 x 8B elements,
// expressed as a 2D tile 16384x1) from global into LDS.  D# bitfields per
// CDNA5 ISA.  Issued per-wave (EXEC ignored).
// This toolchain's builtin is the 6-arg form:
//   (uint32x4 g0, int32x8 g1, int32x4 g2, int32x4 g3, int32x8 extra, i32 cpol)
__device__ __forceinline__ void tdm_load_slice(const void* gsrc,
                                               unsigned lds_byte_addr) {
    unsigned long ga = (unsigned long)gsrc;
    v4u g0;
    g0[0] = 1u;                                   // count=1, user-mode D#
    g0[1] = lds_byte_addr;                        // lds_addr[31:0]
    g0[2] = (unsigned)(ga & 0xffffffffu);         // global_addr[31:0]
    g0[3] = (unsigned)((ga >> 32) & 0x1ffffffu)   // global_addr[56:32]
            | (2u << 30);                         // type=2 ("image")
    v8i g1;
    g1[0] = 3 << 16;                  // workgroup_mask=0, data_size=3 (8B)
    g1[1] = (int)(16384u << 16);      // tensor_dim0[15:0]=16384 in bits 63:48
    g1[2] = 1 << 16;                  // tensor_dim0 hi=0 | tensor_dim1=1
    g1[3] = (int)(16384u << 16);      // tensor_dim1 hi=0 | tile_dim0=16384
    g1[4] = 1;                        // tile_dim1=1, tile_dim2=0
    g1[5] = 16384;                    // tensor_dim0_stride = 16384 elements
    g1[6] = 0;
    g1[7] = 0;
    v4i z4 = {0, 0, 0, 0};            // D# groups 2/3: unused (2-D tensor)
    v8i z8 = {0, 0, 0, 0, 0, 0, 0, 0};
    __builtin_amdgcn_tensor_load_to_lds(g0, g1, z4, z4, z8, 0);
}
#endif

// ---------------------------------------------------------------------------
// K0: pack W_tau / W_f (f32 row-major) into WMMA A-fragment order, bf16.
// Layout: [mat][rt(32)][kt(16)][lane(32)][e(16)]  (1 KB per 16x32 tile).
// Also zeroes the sync counter (must happen every call; stream-ordered).
// ---------------------------------------------------------------------------
__global__ void k_pack(const float* __restrict__ Wt, const float* __restrict__ Wf,
                       __bf16* __restrict__ wpack, unsigned* __restrict__ cnt) {
    unsigned id = blockIdx.x * 256u + threadIdx.x;   // 0 .. 524287
    if (id == 0) *cnt = 0u;
    int e    = id & 15;
    int lane = (id >> 4) & 31;
    int kt   = (id >> 9) & 15;
    int rt   = (id >> 13) & 31;
    int mat  = (id >> 18) & 1;
    int hi   = lane >> 4;
    int row  = rt * 16 + (lane & 15);
    int col  = kt * 32 + e + 16 * hi;
    const float* W = mat ? Wf : Wt;
    wpack[id] = (__bf16)W[row * HIDDEN + col];
}

// ---------------------------------------------------------------------------
// K1: Z[g][n] = U @ I(t_g) + bias, for all 4092 RK4 stage times, as a WMMA
// GEMM: M=4096 (stages, padded), N=512, K=256.  One 16x16 tile per wave.
// ---------------------------------------------------------------------------
__global__ void k_proj(const float* __restrict__ Iseq,
                       const float* __restrict__ Utau, const float* __restrict__ btau,
                       const float* __restrict__ Uf,   const float* __restrict__ bfv,
                       float* __restrict__ Ztau, float* __restrict__ Zf) {
    int wid  = (int)((blockIdx.x * blockDim.x + threadIdx.x) >> 5);
    int lane = threadIdx.x & 31;
    int mat  = wid >> 13;            // 8192 tiles per matrix
    int rem  = wid & 8191;
    int mtile = rem >> 5;            // 0..255
    int ntile = rem & 31;            // 0..31
    int hi = lane >> 4;
    int lm = lane & 15;

    // stage index for this A row; ZOH input lookup (searchsorted-left on
    // linspace(0,1,4096) == ceil(t*4095), clamped)
    int g = mtile * 16 + lm;
    if (g > NSTAGE - 1) g = NSTAGE - 1;
    int s = g >> 2, j = g & 3;
    float frac = (j == 3) ? 1.0f : (float)j * (1.0f / 3.0f);
    float t = ((float)s + frac) * (1.0f / (float)NSTEP);
    int idx = (int)ceilf(t * (float)(T_IN - 1));
    idx = idx < 0 ? 0 : (idx > T_IN - 1 ? T_IN - 1 : idx);
    const float* Irow = Iseq + (long)idx * INPUT;

    const float* U    = mat ? Uf  : Utau;
    const float* bias = mat ? bfv : btau;
    float*       Z    = mat ? Zf  : Ztau;

    int ncol = ntile * 16 + lm;
    v8f acc = {};
    for (int kt = 0; kt < 8; ++kt) {
        v16bf a, b;
        for (int e = 0; e < 16; ++e) {
            a[e] = (__bf16)Irow[kt * 32 + a_kidx(e, hi)];
            // B 32x16 16-bit layout: lane&15 = N, K = a_kidx(e, hi) (same order as A)
            b[e] = (__bf16)U[ncol * INPUT + kt * 32 + a_kidx(e, hi)];
        }
        acc = __builtin_amdgcn_wmma_f32_16x16x32_bf16(false, a, false, b,
                                                      (short)0, acc, false, false);
    }
    float bv = bias[ncol];
    int grow = mtile * 16 + hi * 8;   // C: VGPR r -> M = r + 8*hi, N = lane&15
    for (int r = 0; r < 8; ++r)
        Z[(long)(grow + r) * HIDDEN + ncol] = acc[r] + bv;
}

// ---------------------------------------------------------------------------
// K2: sequential RK4 loop. 4 persistent workgroups; each owns 128 output rows
// and keeps its 256 KB bf16 weight slice resident in LDS (CDNA5: 320 KB/WGP),
// staged via the Tensor Data Mover.  Cross-WG sync: device-scope atomic
// counter (+ s_cluster_barrier, NOP when not cluster-dispatched).
// ---------------------------------------------------------------------------
__global__ __launch_bounds__(256, 1) void k_ode(
        const float* __restrict__ x0, const __bf16* __restrict__ wpack,
        const float* __restrict__ Ztau, const float* __restrict__ Zf,
        const float* __restrict__ tau0, float* __restrict__ xbuf,
        unsigned* __restrict__ cnt, float* __restrict__ out) {
    extern __shared__ char smem[];
    __bf16* wlds = (__bf16*)smem;                    // 131072 elems = 256 KB
    float*  xf32 = (float*)(smem + 262144);          // 512
    __bf16* xbf  = (__bf16*)(smem + 264192);         // 512
    float*  ytau = (float*)(smem + 265216);          // 128
    float*  yfv  = (float*)(smem + 265728);          // 128
    float*  y0l  = (float*)(smem + 266240);          // 128
    float*  k1l  = (float*)(smem + 266752);
    float*  k2l  = (float*)(smem + 267264);
    float*  k3l  = (float*)(smem + 267776);

    const int tid  = threadIdx.x;
    const int b    = blockIdx.x;          // 0..3
    const int lane = tid & 31;
    const int wave = tid >> 5;            // 0..7 -> local row-tile
    const int hi   = lane >> 4;
    const float dt = 1.0f / (float)NSTEP;

    // Stage this WG's weight slice (rows b*128..b*128+127, both mats) into LDS.
#if __has_builtin(__builtin_amdgcn_tensor_load_to_lds)
    if (wave == 0) {
        for (int mat = 0; mat < 2; ++mat)
            tdm_load_slice(wpack + mat * 262144 + b * 65536,
                           (unsigned)(unsigned long)(const void*)(wlds + mat * 65536));
#if __has_builtin(__builtin_amdgcn_s_wait_tensorcnt)
        __builtin_amdgcn_s_wait_tensorcnt(0);
#endif
    }
#else
    for (int mat = 0; mat < 2; ++mat) {
        const uint4* src = (const uint4*)(wpack + mat * 262144 + b * 65536);
        uint4* dst = (uint4*)(wlds + mat * 65536);
        for (int i = tid; i < 8192; i += 256) dst[i] = src[i];
    }
#endif
    // Init state: y0 slice, x ping-pong buffer 0, output row 0.
    if (tid < 128) {
        int grow = b * 128 + tid;
        float v = x0[grow];
        y0l[tid] = v;
        xbuf[grow] = v;
        out[grow] = v;
    }
    __threadfence();
    __syncthreads();
    if (tid == 0)
        __hip_atomic_fetch_add(cnt, 1u, __ATOMIC_RELEASE, __HIP_MEMORY_SCOPE_AGENT);

    for (int g = 0; g < NSTAGE; ++g) {
        // Wait until all 4 WGs published their slice of x_g.
        if (tid == 0) {
            unsigned target = (unsigned)NWG * (unsigned)(g + 1);
            while (__hip_atomic_load(cnt, __ATOMIC_ACQUIRE,
                                     __HIP_MEMORY_SCOPE_AGENT) < target)
                __builtin_amdgcn_s_sleep(1);
        }
        __syncthreads();
        __builtin_amdgcn_s_cluster_barrier();   // NOP unless cluster dispatch

        // Pull full x_g (2 KB from L2) into LDS, f32 master + bf16 copy.
        {
            const float* xg = xbuf + (g & 1) * HIDDEN;
            int i0 = tid * 2;
            float v0 = xg[i0], v1 = xg[i0 + 1];
            xf32[i0] = v0; xf32[i0 + 1] = v1;
            xbf[i0] = (__bf16)v0; xbf[i0 + 1] = (__bf16)v1;
        }
        __syncthreads();

        // Matvec slices via WMMA: wave w handles row-tile w for both matrices.
        {
            const int rt = wave;
            for (int mat = 0; mat < 2; ++mat) {
                v8f acc = {};
                const __bf16* wbase = wlds + ((mat * 8 + rt) * 16) * 512;
                for (int kt = 0; kt < 16; ++kt) {
                    v16bf a = *(const v16bf*)(wbase + kt * 512 + lane * 16);
                    v16bf bfrag{};
                    if ((lane & 15) == 0)     // only column N=0 carries x
                        bfrag = *(const v16bf*)(xbf + kt * 32 + hi * 16);
                    acc = __builtin_amdgcn_wmma_f32_16x16x32_bf16(
                              false, a, false, bfrag, (short)0, acc, false, false);
                }
                if ((lane & 15) == 0) {       // C col 0 lives in lanes 0 / 16
                    float* yv = mat ? yfv : ytau;
                    int base = rt * 16 + hi * 8;
                    for (int r = 0; r < 8; ++r) yv[base + r] = acc[r];
                }
            }
        }
        __syncthreads();

        // Elementwise dyn + RK4 (Kutta 3/8) combine; publish next-x slice.
        if (tid < 128) {
            int grow = b * 128 + tid;
            long zoff = (long)g * HIDDEN + grow;
            float vt = ytau[tid] + Ztau[zoff];
            float vf = yfv[tid]  + Zf[zoff];
            __builtin_prefetch(Ztau + zoff + HIDDEN, 0, 1);
            __builtin_prefetch(Zf   + zoff + HIDDEN, 0, 1);
            float sp  = fmaxf(vt, 0.0f) + log1pf(expf(-fabsf(vt)));
            float tau = tau0[grow] + sp;
            float fv  = tanhf(vf);
            float k   = -xf32[grow] / tau + fv;

            int j = g & 3;
            float y0v = y0l[tid];
            float xn;
            if (j == 0)      { k1l[tid] = k; xn = y0v + dt * (1.0f / 3.0f) * k; }
            else if (j == 1) { k2l[tid] = k; xn = y0v + dt * (k - k1l[tid] * (1.0f / 3.0f)); }
            else if (j == 2) { k3l[tid] = k; xn = y0v + dt * (k1l[tid] - k2l[tid] + k); }
            else {
                float y1 = y0v + (k1l[tid] + 3.0f * (k2l[tid] + k3l[tid]) + k) * dt * 0.125f;
                xn = y1; y0l[tid] = y1;
                int step = g >> 2;
                out[(long)(step + 1) * HIDDEN + grow] = y1;
            }
            xbuf[((g + 1) & 1) * HIDDEN + grow] = xn;
        }
        __threadfence();
        __syncthreads();
        if (tid == 0)
            __hip_atomic_fetch_add(cnt, 1u, __ATOMIC_RELEASE, __HIP_MEMORY_SCOPE_AGENT);
    }
}

// ---------------------------------------------------------------------------
extern "C" void kernel_launch(void* const* d_in, const int* in_sizes, int n_in,
                              void* d_out, int out_size, void* d_ws, size_t ws_size,
                              hipStream_t stream) {
    (void)in_sizes; (void)n_in; (void)out_size; (void)ws_size;
    const float* x0   = (const float*)d_in[0];
    const float* Iseq = (const float*)d_in[1];
    /* d_in[2] = t_save: uniform linspace(0,1,1024) -> dt = 1/1023 constant */
    const float* Wtau = (const float*)d_in[3];
    const float* Utau = (const float*)d_in[4];
    const float* btau = (const float*)d_in[5];
    const float* tau0 = (const float*)d_in[6];
    const float* Wf   = (const float*)d_in[7];
    const float* Uf   = (const float*)d_in[8];
    const float* bfv  = (const float*)d_in[9];

    char* ws = (char*)d_ws;
    unsigned* cnt  = (unsigned*)ws;
    float*    xbuf = (float*)(ws + WS_XBUF);
    __bf16*   wpk  = (__bf16*)(ws + WS_WPACK);
    float*    Ztau = (float*)(ws + WS_ZTAU);
    float*    Zf   = (float*)(ws + WS_ZF);

    // K0: pack weights to bf16 WMMA fragments + reset sync counter.
    k_pack<<<2048, 256, 0, stream>>>(Wtau, Wf, wpk, cnt);
    // K1: all 4092 input projections as one full-utilization WMMA GEMM.
    k_proj<<<2048, 256, 0, stream>>>(Iseq, Utau, btau, Uf, bfv, Ztau, Zf);
    // K2: sequential RK4 with LDS-resident weights across 4 WGPs.
    k_ode<<<NWG, 256, 268288, stream>>>(x0, wpk, Ztau, Zf, tau0, xbuf, cnt,
                                        (float*)d_out);
}
